// CrossAttention_48026324303962
// MI455X (gfx1250) — matrix operand order repaired and checked by reading, not compiled, not run on hardware
//
#include <hip/hip_runtime.h>

typedef __attribute__((ext_vector_type(4)))  _Float16 v4h;
typedef __attribute__((ext_vector_type(8)))  _Float16 v8h;
typedef __attribute__((ext_vector_type(16))) _Float16 v16h;
typedef __attribute__((ext_vector_type(8)))  float    v8f;

#define ATT_H 16
#define ATT_D 64

// ---------- WMMA fragment loaders (layouts per CDNA5 ISA §7.12.2) ----------

// A-matrix 16x32 f16: lane = row (L&15); lanes 0-15 hold K={0..7,16..23},
// lanes 16-31 hold K={8..15,24..31}. Two contiguous 8-f16 (16B) runs per lane.
__device__ __forceinline__ v16h load_a_frag(const _Float16* base, int ld) {
  const int lane = threadIdx.x & 31;
  const int row  = lane & 15;
  const int kb   = (lane >> 4) << 3;           // 0 or 8
  const _Float16* p = base + (size_t)row * ld + kb;
  v8h lo = *(const v8h*)(p);
  v8h hi = *(const v8h*)(p + 16);
  v16h a;
#pragma unroll
  for (int i = 0; i < 8; ++i) { a[i] = lo[i]; a[i + 8] = hi[i]; }
  return a;
}

// B-matrix 32x16 f16 where B[k][n] = W[n][k] (W row-major, K contiguous):
// lane n = L&15; lanes 0-15 hold K={0..7,16..23}, lanes 16-31 hold K={8..15,24..31}.
// Two contiguous 8-f16 (16B) runs per lane.
__device__ __forceinline__ v16h load_b_frag(const _Float16* base, int ld) {
  const int lane = threadIdx.x & 31;
  const int n    = lane & 15;
  const int kb   = (lane >> 4) << 3;           // 0 or 8
  const _Float16* p = base + (size_t)n * ld + kb;
  v8h lo = *(const v8h*)(p);
  v8h hi = *(const v8h*)(p + 16);
  v16h b;
#pragma unroll
  for (int i = 0; i < 8; ++i) { b[i] = lo[i]; b[i + 8] = hi[i]; }
  return b;
}

// Async Global -> LDS copy, 16 bytes per lane (ASYNCcnt-tracked, GV mode).
__device__ __forceinline__ void async_load_b128(unsigned lds_off, const void* gaddr) {
  asm volatile("global_load_async_to_lds_b128 %0, %1, off"
               :: "v"(lds_off), "v"(gaddr)
               : "memory");
}
__device__ __forceinline__ void wait_async0() {
  asm volatile("s_wait_asynccnt 0" ::: "memory");
}

// ---------- f32 -> f16 conversion ----------
__global__ void __launch_bounds__(256)
cvt_f32_f16(const float* __restrict__ in, _Float16* __restrict__ out, int n4) {
  int i = blockIdx.x * 256 + threadIdx.x;
  if (i < n4) {
    float4 v = ((const float4*)in)[i];
    v4h o;
    o[0] = (_Float16)v.x; o[1] = (_Float16)v.y;
    o[2] = (_Float16)v.z; o[3] = (_Float16)v.w;
    ((v4h*)out)[i] = o;
  }
}

// ---------- Generic GEMM: out(M,N) = A(MxK,row-major) * W(NxK,row-major)^T ----------
// Block = 256 thr = 8 waves; block tile 256(M) x 64(N); wave tile 32x64
// (2 A-frags x 4 B-frags = 8 WMMA per 32-wide k-step).
// B-tile (64x32 f16 = 4KB) is staged in LDS via async global->LDS loads,
// double-buffered: tile k+1 streams in while tile k is consumed.
// MODE 0: store f16 row-major.  MODE 1: store f16 transposed (B,H,D,Lkv) for V.
// MODE 2: store f32 row-major + bias (final projection to d_out).
template <int MODE>
__global__ void __launch_bounds__(256)
gemm_wmma(const _Float16* __restrict__ A, int lda,
          const _Float16* __restrict__ W, int ldw,
          void* __restrict__ outp, int ldo, int K,
          const float* __restrict__ bias, int rows_per_batch) {
  const int tid  = threadIdx.x;
  const int wave = tid >> 5;
  const int lane = tid & 31;
  const int m0 = blockIdx.x * 256 + wave * 32;
  const int n0 = blockIdx.y * 64;

  __shared__ __align__(16) _Float16 bt[2][64 * 32];  // 2 x 4KB B-tiles

  // cooperative async fetch: thread t brings 8 f16: row n = t/4, k = (t%4)*8
  const int tn = tid >> 2;
  const int tk = (tid & 3) << 3;
  const _Float16* wrow = W + (size_t)(n0 + tn) * ldw + tk;
  const unsigned lds0 = (unsigned)(size_t)(const void*)&bt[0][0] + (unsigned)(tid * 16);
  const unsigned lds_stride = 64u * 32u * 2u;        // 4KB between buffers

  // prologue: stage tile for k0 = 0 into buffer 0
  async_load_b128(lds0, (const void*)wrow);
  wait_async0();
  __syncthreads();

  v8f acc[2][4] = {};
  const _Float16* aptr0 = A + (size_t)m0 * lda;
  const _Float16* aptr1 = A + (size_t)(m0 + 16) * lda;

  int buf = 0;
  for (int k0 = 0; k0 < K; k0 += 32) {
    if (k0 + 32 < K)  // prefetch next B-tile into the other buffer
      async_load_b128(lds0 + (unsigned)(buf ^ 1) * lds_stride,
                      (const void*)(wrow + k0 + 32));

    v16h a0 = load_a_frag(aptr0 + k0, lda);
    v16h a1 = load_a_frag(aptr1 + k0, lda);
    const _Float16* bbase = &bt[buf][0];
#pragma unroll
    for (int t = 0; t < 4; ++t) {
      v16h b = load_b_frag(bbase + (size_t)(16 * t) * 32, 32);
      acc[0][t] = __builtin_amdgcn_wmma_f32_16x16x32_f16(
          false, a0, false, b, (short)0, acc[0][t], false, false);
      acc[1][t] = __builtin_amdgcn_wmma_f32_16x16x32_f16(
          false, a1, false, b, (short)0, acc[1][t], false, false);
    }
    wait_async0();       // next tile landed
    __syncthreads();     // everyone done reading current tile
    buf ^= 1;
  }

  const int nl = lane & 15;
  const int half = lane >> 4;
#pragma unroll
  for (int sub = 0; sub < 2; ++sub) {
#pragma unroll
    for (int t = 0; t < 4; ++t) {
      const int n = n0 + 16 * t + nl;
#pragma unroll
      for (int r = 0; r < 8; ++r) {
        const int m = m0 + 16 * sub + r + 8 * half;
        const float v = acc[sub][t][r];
        if (MODE == 0) {
          ((_Float16*)outp)[(size_t)m * ldo + n] = (_Float16)v;
        } else if (MODE == 1) {
          const int bb = m / rows_per_batch;
          const int ll = m - bb * rows_per_batch;
          const int hh = n >> 6;
          const int dd = n & 63;
          ((_Float16*)outp)[(((size_t)(bb * ATT_H + hh)) * ATT_D + dd) *
                                (size_t)rows_per_batch + ll] = (_Float16)v;
        } else {
          ((float*)outp)[(size_t)m * ldo + n] = v + bias[n];
        }
      }
    }
  }
}

// ---------- Flash attention: per-wave 16 q-rows, kv tiles of 32, online softmax ----------
__global__ void __launch_bounds__(256)
attn_wmma(const _Float16* __restrict__ qh,   // (B,Lq,C) f16, head chunks of 64 along C
          const _Float16* __restrict__ kk,   // (B,Lkv,C) f16
          const _Float16* __restrict__ vT,   // (B,H,D,Lkv) f16
          const float* __restrict__ pos,     // (B,H,Lq,Lpos) f32
          _Float16* __restrict__ xout,       // (B,Lq,C) f16
          int Lq, int Lkv, int Lpos, int C) {
  const int wave = threadIdx.x >> 5;
  const int lane = threadIdx.x & 31;
  const int nl = lane & 15;
  const int half = lane >> 4;

  const int qtiles = Lq / 128;
  const int bh = blockIdx.x / qtiles;
  const int b = bh / ATT_H;
  const int h = bh % ATT_H;
  const int q0 = (blockIdx.x % qtiles) * 128 + wave * 16;

  const _Float16* qbase = qh + ((size_t)b * Lq + q0) * C + h * ATT_D;
  const _Float16* kbase = kk + (size_t)b * Lkv * C + h * ATT_D;
  const _Float16* vbase = vT + ((size_t)(b * ATT_H + h)) * ATT_D * Lkv;
  const float* pbase = pos + (((size_t)(b * ATT_H + h)) * Lq + q0) * (size_t)Lpos;

  const float scale = 0.125f;  // D^-0.5, D=64

  // Q fragments for both 32-wide k-steps of the D=64 dot product (kept in regs)
  const v16h qa0 = load_a_frag(qbase, C);
  const v16h qa1 = load_a_frag(qbase + 32, C);

  float mrow[8], lrow[8];
#pragma unroll
  for (int r = 0; r < 8; ++r) { mrow[r] = -3.0e38f; lrow[r] = 0.0f; }
  v8f o[4] = {};

  __shared__ __align__(32) _Float16 pshare[8][16 * 32];  // per-wave P buffer
  _Float16* pw = &pshare[wave][0];

  for (int kv0 = 0; kv0 < Lkv; kv0 += 32) {
    // ---- scores S = Q K^T for 16x32 kv block (2 tiles x 2 k-steps) ----
    v8f s0 = {}, s1 = {};
    {
      v16h kb0 = load_b_frag(kbase + (size_t)kv0 * C, C);
      v16h kb1 = load_b_frag(kbase + (size_t)kv0 * C + 32, C);
      s0 = __builtin_amdgcn_wmma_f32_16x16x32_f16(false, qa0, false, kb0, (short)0, s0, false, false);
      s0 = __builtin_amdgcn_wmma_f32_16x16x32_f16(false, qa1, false, kb1, (short)0, s0, false, false);
      v16h kb2 = load_b_frag(kbase + (size_t)(kv0 + 16) * C, C);
      v16h kb3 = load_b_frag(kbase + (size_t)(kv0 + 16) * C + 32, C);
      s1 = __builtin_amdgcn_wmma_f32_16x16x32_f16(false, qa0, false, kb2, (short)0, s1, false, false);
      s1 = __builtin_amdgcn_wmma_f32_16x16x32_f16(false, qa1, false, kb3, (short)0, s1, false, false);
    }

    // ---- scale + positional bias (pos is zero-padded beyond Lpos) ----
    const int kvA = kv0 + nl;
    const int kvB = kv0 + 16 + nl;
#pragma unroll
    for (int r = 0; r < 8; ++r) {
      const int qr = r + 8 * half;  // C/D layout: VGPR r holds row r (+8 for upper half)
      float p0 = (kvA < Lpos) ? pbase[(size_t)qr * Lpos + kvA] : 0.0f;
      float p1 = (kvB < Lpos) ? pbase[(size_t)qr * Lpos + kvB] : 0.0f;
      s0[r] = s0[r] * scale + p0;
      s1[r] = s1[r] * scale + p1;
    }

    // ---- online softmax: row max over the 16-lane half holding this row ----
    float mt[8];
#pragma unroll
    for (int r = 0; r < 8; ++r) mt[r] = fmaxf(s0[r], s1[r]);
#pragma unroll
    for (int mask = 1; mask <= 8; mask <<= 1)
#pragma unroll
      for (int r = 0; r < 8; ++r) mt[r] = fmaxf(mt[r], __shfl_xor(mt[r], mask, 32));

    float corr[8], rs[8];
#pragma unroll
    for (int r = 0; r < 8; ++r) {
      const float mn = fmaxf(mrow[r], mt[r]);
      corr[r] = __expf(mrow[r] - mn);
      mrow[r] = mn;
      const float e0 = __expf(s0[r] - mn);
      const float e1 = __expf(s1[r] - mn);
      s0[r] = e0; s1[r] = e1;
      rs[r] = e0 + e1;
    }
#pragma unroll
    for (int mask = 1; mask <= 8; mask <<= 1)
#pragma unroll
      for (int r = 0; r < 8; ++r) rs[r] += __shfl_xor(rs[r], mask, 32);
#pragma unroll
    for (int r = 0; r < 8; ++r) lrow[r] = lrow[r] * corr[r] + rs[r];
#pragma unroll
    for (int t = 0; t < 4; ++t)
#pragma unroll
      for (int r = 0; r < 8; ++r) o[t][r] *= corr[r];

    // ---- P (C/D layout) -> LDS row-major 16x32 f16 -> A-fragment ----
#pragma unroll
    for (int r = 0; r < 8; ++r) {
      const int qr = r + 8 * half;
      pw[qr * 32 + nl]      = (_Float16)s0[r];
      pw[qr * 32 + 16 + nl] = (_Float16)s1[r];
    }
    asm volatile("s_wait_dscnt 0" ::: "memory");  // wave-private buffer: DS order only
    const v16h pa = load_a_frag(pw, 32);

    // ---- O += P * V  (V^T gives contiguous B-fragments) ----
#pragma unroll
    for (int t = 0; t < 4; ++t) {
      v16h vb = load_b_frag(vbase + (size_t)(16 * t) * Lkv + kv0, Lkv);
      o[t] = __builtin_amdgcn_wmma_f32_16x16x32_f16(false, pa, false, vb, (short)0, o[t], false, false);
    }
  }

  // ---- normalize and store (B,Lq,C) f16 ----
#pragma unroll
  for (int r = 0; r < 8; ++r) {
    const float inv = 1.0f / lrow[r];
    const int m = q0 + r + 8 * half;
#pragma unroll
    for (int t = 0; t < 4; ++t) {
      xout[((size_t)b * Lq + m) * C + h * ATT_D + 16 * t + nl] =
          (_Float16)(o[t][r] * inv);
    }
  }
}

// ---------- host launcher ----------
extern "C" void kernel_launch(void* const* d_in, const int* in_sizes, int n_in,
                              void* d_out, int out_size, void* d_ws, size_t ws_size,
                              hipStream_t stream) {
  const float* q    = (const float*)d_in[0];
  const float* kv   = (const float*)d_in[1];
  const float* pos  = (const float*)d_in[2];
  const float* q_w  = (const float*)d_in[3];
  const float* kv_w = (const float*)d_in[4];
  const float* p_w  = (const float*)d_in[5];
  const float* p_b  = (const float*)d_in[6];

  const int B = 4, Lq = 1024, Lkv = 2048, C = 1024, Lpos = 1024;

  char* ws = (char*)d_ws;
  _Float16* q16   = (_Float16*)(ws + (size_t)0);          //  8 MB  (B,Lq,C)
  _Float16* kv16  = (_Float16*)(ws + ((size_t)8  << 20)); // 16 MB  (B,Lkv,C)
  _Float16* qw16  = (_Float16*)(ws + ((size_t)24 << 20)); //  2 MB  (C,C)
  _Float16* kvw16 = (_Float16*)(ws + ((size_t)26 << 20)); //  4 MB  (2C,C)
  _Float16* pw16  = (_Float16*)(ws + ((size_t)30 << 20)); //  2 MB  (C,C)
  _Float16* qh16  = (_Float16*)(ws + ((size_t)32 << 20)); //  8 MB  (B,Lq,C)
  _Float16* k16   = (_Float16*)(ws + ((size_t)40 << 20)); // 16 MB  (B,Lkv,C)
  _Float16* vT16  = (_Float16*)(ws + ((size_t)56 << 20)); // 16 MB  (B,H,D,Lkv)
  _Float16* x16   = (_Float16*)(ws + ((size_t)72 << 20)); //  8 MB  (B,Lq,C)

  auto cvt = [&](const float* src, _Float16* dst, size_t n) {
    const int n4 = (int)(n / 4);
    cvt_f32_f16<<<dim3((n4 + 255) / 256), dim3(256), 0, stream>>>(src, dst, n4);
  };
  cvt(q,    q16,   (size_t)B * Lq * C);
  cvt(kv,   kv16,  (size_t)B * Lkv * C);
  cvt(q_w,  qw16,  (size_t)C * C);
  cvt(kv_w, kvw16, (size_t)2 * C * C);
  cvt(p_w,  pw16,  (size_t)C * C);

  // Q projection: (B*Lq, C) = q16 * q_w^T
  gemm_wmma<0><<<dim3((B * Lq) / 256, C / 64), dim3(256), 0, stream>>>(
      q16, C, qw16, C, (void*)qh16, C, C, nullptr, 0);
  // K projection: (B*Lkv, C) = kv16 * kv_w[0:C]^T
  gemm_wmma<0><<<dim3((B * Lkv) / 256, C / 64), dim3(256), 0, stream>>>(
      kv16, C, kvw16, C, (void*)k16, C, C, nullptr, 0);
  // V projection with transposed store -> (B,H,D,Lkv)
  gemm_wmma<1><<<dim3((B * Lkv) / 256, C / 64), dim3(256), 0, stream>>>(
      kv16, C, kvw16 + (size_t)C * C, C, (void*)vT16, C, C, nullptr, Lkv);
  // Flash attention
  attn_wmma<<<dim3(B * ATT_H * (Lq / 128)), dim3(256), 0, stream>>>(
      qh16, k16, vT16, pos, x16, Lq, Lkv, Lpos, C);
  // Output projection + bias -> f32 d_out
  gemm_wmma<2><<<dim3((B * Lq) / 256, C / 64), dim3(256), 0, stream>>>(
      x16, C, pw16, C, d_out, C, C, p_b, 0);
}
